// NeuralLogicLayer_9010841387133
// MI455X (gfx1250) — matrix operand order repaired and checked by reading, not compiled, not run on hardware
//
#include <hip/hip_runtime.h>
#include <hip/hip_bf16.h>

// ---------------------------------------------------------------------------
// B=16, N=128, C=64, H=128
//   order-2: M = 262144 rows, K1=256 -> H=128 -> 64
//   order-1: M = 2048   rows, K1=256
//   order-0: M = 16     rows, K1=192
// Strategy: f16 WMMA (16x16x32, f32 accum).  Weights pre-transposed to
// WT[N][K] f16 and staged per-block into LDS (80KB set overflows the 64KB
// WGP$ but fits the 320KB LDS).  Each wave owns MT M-tiles so every
// B-fragment ds_load feeds MT WMMAs.
// ---------------------------------------------------------------------------

typedef __attribute__((ext_vector_type(16))) _Float16 v16h;
typedef __attribute__((ext_vector_type(8)))  _Float16 v8h;
typedef __attribute__((ext_vector_type(8)))  float    v8f;
typedef __attribute__((ext_vector_type(4)))  float    v4f;

__device__ __forceinline__ v8f wmma_f16(v16h a, v16h b, v8f c) {
  return __builtin_amdgcn_wmma_f32_16x16x32_f16(
      false, a, false, b, (short)0, c, false, false);
}

// A-fragment (16x32 f16): halves[0..7] = row[k0..k0+7], halves[8..15] =
// row[k0+16..k0+23], k0 = ktile*32 + (lane>>4)*8 -> two 16B loads.
__device__ __forceinline__ v16h ldA16(const _Float16* p0, const _Float16* p1) {
  v8h a = *(const v8h*)p0;
  v8h b = *(const v8h*)p1;
  v16h r;
#pragma unroll
  for (int t = 0; t < 8; ++t) { r[t] = a[t]; r[8 + t] = b[t]; }
  return r;
}

// Same fragment gathered from f32 rows with on-the-fly convert.
__device__ __forceinline__ v16h ldA32(const float* p0, const float* p1) {
  v4f a0 = *(const v4f*)p0;
  v4f a1 = *(const v4f*)(p0 + 4);
  v4f b0 = *(const v4f*)p1;
  v4f b1 = *(const v4f*)(p1 + 4);
  v16h r;
#pragma unroll
  for (int t = 0; t < 4; ++t) {
    r[t]      = (_Float16)a0[t];
    r[4 + t]  = (_Float16)a1[t];
    r[8 + t]  = (_Float16)b0[t];
    r[12 + t] = (_Float16)b1[t];
  }
  return r;
}

// B-fragment (32x16 f16): lane n = lane&15 holds K k0..k0+7 and
// k0+16..k0+23, k0 = ktile*32 + (lane>>4)*8; WT[n][k] row-major -> two 16B loads.
__device__ __forceinline__ v16h ldB16(const _Float16* p) {
  v8h a = *(const v8h*)p;
  v8h b = *(const v8h*)(p + 16);
  v16h r;
#pragma unroll
  for (int t = 0; t < 8; ++t) { r[t] = a[t]; r[8 + t] = b[t]; }
  return r;
}

// Cooperative copy of the (repacked f16) weights into LDS, then barrier.
__device__ __forceinline__ void stage_weights(const _Float16* ga, int na,
                                              const _Float16* gb, int nb,
                                              _Float16* la, _Float16* lb) {
  for (int idx = threadIdx.x; idx < (na >> 3); idx += (int)blockDim.x)
    ((uint4*)la)[idx] = ((const uint4*)ga)[idx];
  for (int idx = threadIdx.x; idx < (nb >> 3); idx += (int)blockDim.x)
    ((uint4*)lb)[idx] = ((const uint4*)gb)[idx];
  __syncthreads();
}

// ---------------------------------------------------------------------------
// Two-layer MLP over MT 16-row tiles per wave, weights in LDS.
//   layer A: [MT*16 x K1] @ Wa^T -> relu -> h (LDS relayout)
//   layer B: h @ Wb^T -> sigmoid -> out
// ---------------------------------------------------------------------------
template <int NK1, int MT, typename AL>
__device__ __forceinline__ void run_mlp(
    AL loadA,                                 // (mt, kt) -> v16h
    const _Float16* lwa, const float* __restrict__ ba,
    const _Float16* lwb, const float* __restrict__ bb,
    float* __restrict__ outBase,              // rows row0 .. row0+MT*16
    _Float16* hb)                             // per-wave LDS [MT][16][128]
{
  constexpr int K1 = NK1 * 32;
  const int lane = threadIdx.x & 31;
  const int g    = lane >> 4;
  const int nlo  = lane & 15;

  // ---- A-fragments for all MT tiles: resident for the whole of layer A
  v16h afr[MT][NK1];
#pragma unroll
  for (int mt = 0; mt < MT; ++mt)
#pragma unroll
    for (int kt = 0; kt < NK1; ++kt) afr[mt][kt] = loadA(mt, kt);

  // ---- layer A: each LDS B-fragment feeds MT WMMAs
#pragma unroll
  for (int nt = 0; nt < 8; ++nt) {
    v8f acc[MT];
#pragma unroll
    for (int mt = 0; mt < MT; ++mt) acc[mt] = (v8f){};
#pragma unroll
    for (int kt = 0; kt < NK1; ++kt) {
      v16h bf = ldB16(lwa + (nt * 16 + nlo) * K1 + kt * 32 + g * 8);
#pragma unroll
      for (int mt = 0; mt < MT; ++mt)
        acc[mt] = wmma_f16(afr[mt][kt], bf, acc[mt]);
    }
    const float bias = ba[nt * 16 + nlo];
#pragma unroll
    for (int mt = 0; mt < MT; ++mt)
#pragma unroll
      for (int r = 0; r < 8; ++r) {
        float h = acc[mt][r] + bias;
        h = h > 0.f ? h : 0.f;
        hb[(mt * 16 + r + g * 8) * 128 + nt * 16 + nlo] = (_Float16)h;
      }
  }

  // same-wave LDS RAW across the h relayout
  asm volatile("s_wait_dscnt 0" ::: "memory");

  // ---- layer B: h as A-fragments (afr now dead -> pressure capped)
  v16h hfr[MT][4];
#pragma unroll
  for (int mt = 0; mt < MT; ++mt)
#pragma unroll
    for (int kt = 0; kt < 4; ++kt) {
      const _Float16* p = hb + (mt * 16 + nlo) * 128 + kt * 32 + g * 8;
      hfr[mt][kt] = ldA16(p, p + 16);
    }

#pragma unroll
  for (int nt = 0; nt < 4; ++nt) {
    v8f acc[MT];
#pragma unroll
    for (int mt = 0; mt < MT; ++mt) acc[mt] = (v8f){};
#pragma unroll
    for (int kt = 0; kt < 4; ++kt) {
      v16h bf = ldB16(lwb + (nt * 16 + nlo) * 128 + kt * 32 + g * 8);
#pragma unroll
      for (int mt = 0; mt < MT; ++mt)
        acc[mt] = wmma_f16(hfr[mt][kt], bf, acc[mt]);
    }
    const float bias = bb[nt * 16 + nlo];
#pragma unroll
    for (int mt = 0; mt < MT; ++mt)
#pragma unroll
      for (int r = 0; r < 8; ++r) {
        float xv = acc[mt][r] + bias;
        outBase[(mt * 16 + r + g * 8) * 64 + nt * 16 + nlo] =
            1.f / (1.f + __expf(-xv));
      }
  }
}

// ---------------------------------------------------------------------------
// Order-2: block = 8 waves x (MT=2 tiles) = 256 rows; x2 rows gathered on
// the fly.  Row r = ((b*128)+i)*128 + j.
// ---------------------------------------------------------------------------
__global__ __launch_bounds__(256) void k_order2(
    const float* __restrict__ f1, const float* __restrict__ f2,
    const _Float16* __restrict__ w2aT, const float* __restrict__ b2a,
    const _Float16* __restrict__ w2bT, const float* __restrict__ b2b,
    float* __restrict__ out2) {
  __shared__ __align__(16) _Float16 lwa[128 * 256];      // 64 KB
  __shared__ __align__(16) _Float16 lwb[64 * 128];       // 16 KB
  __shared__ __align__(16) _Float16 hb[8 * 2 * 16 * 128];// 64 KB
  stage_weights(w2aT, 128 * 256, w2bT, 64 * 128, lwa, lwb);

  const int wave = threadIdx.x >> 5;
  const int lane = threadIdx.x & 31;
  const int g    = lane >> 4;
  const int m    = lane & 15;

  const long row0 = (long)blockIdx.x * 256 + wave * 32;

  // x2 row = [ f1[b,i] | f2[b,i,j] | f1[b,j] | f2[b,j,i] ]
  const float* sg[2][4];
#pragma unroll
  for (int mt = 0; mt < 2; ++mt) {
    const long gr = row0 + mt * 16 + m;
    const int  b  = (int)(gr >> 14);
    const int  i  = (int)((gr >> 7) & 127);
    const int  j  = (int)(gr & 127);
    sg[mt][0] = f1 + ((long)b * 128 + i) * 64;
    sg[mt][1] = f2 + gr * 64;
    sg[mt][2] = f1 + ((long)b * 128 + j) * 64;
    sg[mt][3] = f2 + (((long)b * 128 + j) * 128 + i) * 64;
  }

  auto loadA = [&](int mt, int kt) -> v16h {
    const float* s = sg[mt][kt >> 1];
    const int base = (kt & 1) * 32 + g * 8;
    return ldA32(s + base, s + base + 16);
  };

  run_mlp<8, 2>(loadA, lwa, b2a, lwb, b2b, out2 + row0 * 64,
                hb + wave * (2 * 16 * 128));
}

// ---------------------------------------------------------------------------
// Generic kernel for order-0 / order-1 (activations pre-built f16 rows).
// ---------------------------------------------------------------------------
template <int NK1, int MT>
__global__ __launch_bounds__(256) void k_mlp_ws(
    const _Float16* __restrict__ xws,
    const _Float16* __restrict__ waT, const float* __restrict__ ba,
    const _Float16* __restrict__ wbT, const float* __restrict__ bb,
    float* __restrict__ out, int Mtasks) {
  __shared__ __align__(16) _Float16 lwa[128 * NK1 * 32];
  __shared__ __align__(16) _Float16 lwb[64 * 128];
  __shared__ __align__(16) _Float16 hb[8 * MT * 16 * 128];
  stage_weights(waT, 128 * NK1 * 32, wbT, 64 * 128, lwa, lwb);

  const int wave = threadIdx.x >> 5;
  const int task = blockIdx.x * 8 + wave;
  if (task >= Mtasks) return;  // wave-uniform: EXEC stays all-ones for WMMA
  const int lane = threadIdx.x & 31;
  const int g    = lane >> 4;
  const int m    = lane & 15;

  const long row0 = (long)task * (MT * 16);

  auto loadA = [&](int mt, int kt) -> v16h {
    const _Float16* p =
        xws + (row0 + mt * 16 + m) * (NK1 * 32) + kt * 32 + g * 8;
    return ldA16(p, p + 16);
  };

  run_mlp<NK1, MT>(loadA, lwa, ba, lwb, bb, out + row0 * 64,
                   hb + wave * (MT * 16 * 128));
}

// ---------------------------------------------------------------------------
// Prep kernels
// ---------------------------------------------------------------------------
__global__ void k_repackT(const float* __restrict__ W, _Float16* __restrict__ wT,
                          int K, int N) {
  int t = blockIdx.x * blockDim.x + threadIdx.x;
  if (t >= K * N) return;
  int k = t / N, n = t - k * N;
  wT[n * K + k] = (_Float16)W[(long)k * N + n];
}

__global__ void k_build_x0(const float* __restrict__ f0,
                           const float* __restrict__ f1,
                           _Float16* __restrict__ x0ws) {
  const int b = blockIdx.x;   // 16
  const int c = threadIdx.x;  // 64
  float mx = -1e30f, mn = 1e30f;
  for (int i = 0; i < 128; ++i) {
    float v = f1[((long)b * 128 + i) * 64 + c];
    mx = fmaxf(mx, v);
    mn = fminf(mn, v);
  }
  _Float16* row = x0ws + b * 192;
  row[c]          = (_Float16)f0[b * 64 + c];
  row[64 + 2 * c] = (_Float16)mx;
  row[65 + 2 * c] = (_Float16)mn;
}

__global__ void k_build_x1(const float* __restrict__ f0,
                           const float* __restrict__ f1,
                           const float* __restrict__ f2,
                           _Float16* __restrict__ x1ws) {
  const long bi = blockIdx.x;  // b*128 + i
  const int b = (int)(bi >> 7);
  const int i = (int)(bi & 127);
  const int c = threadIdx.x;   // 64
  float mx = 0.0f, mn = 1.0f;  // reference's masked fill values
  const float* base = f2 + (bi * 128) * 64 + c;
  for (int k = 0; k < 128; ++k) {
    if (k == i) continue;
    float v = base[(long)k * 64];
    mx = fmaxf(mx, v);
    mn = fminf(mn, v);
  }
  _Float16* row = x1ws + bi * 256;
  row[c]           = (_Float16)f0[b * 64 + c];
  row[64 + c]      = (_Float16)f1[bi * 64 + c];
  row[128 + 2 * c] = (_Float16)mx;
  row[129 + 2 * c] = (_Float16)mn;
}

// ---------------------------------------------------------------------------
extern "C" void kernel_launch(void* const* d_in, const int* in_sizes, int n_in,
                              void* d_out, int out_size, void* d_ws,
                              size_t ws_size, hipStream_t stream) {
  (void)in_sizes; (void)n_in; (void)out_size; (void)ws_size;

  const float* f0  = (const float*)d_in[0];
  const float* f1  = (const float*)d_in[1];
  const float* f2  = (const float*)d_in[2];
  const float* W0a = (const float*)d_in[3];
  const float* b0a = (const float*)d_in[4];
  const float* W0b = (const float*)d_in[5];
  const float* b0b = (const float*)d_in[6];
  const float* W1a = (const float*)d_in[7];
  const float* b1a = (const float*)d_in[8];
  const float* W1b = (const float*)d_in[9];
  const float* b1b = (const float*)d_in[10];
  const float* W2a = (const float*)d_in[11];
  const float* b2a = (const float*)d_in[12];
  const float* W2b = (const float*)d_in[13];
  const float* b2b = (const float*)d_in[14];

  float* out0 = (float*)d_out;
  float* out1 = out0 + 16 * 64;
  float* out2 = out1 + 16 * 128 * 64;

  _Float16* ws   = (_Float16*)d_ws;
  _Float16* w0aT = ws;                      // 128*192
  _Float16* w1aT = w0aT + 128 * 192;        // 128*256
  _Float16* w2aT = w1aT + 128 * 256;        // 128*256
  _Float16* w0bT = w2aT + 128 * 256;        // 64*128
  _Float16* w1bT = w0bT + 64 * 128;
  _Float16* w2bT = w1bT + 64 * 128;
  _Float16* x0ws = w2bT + 64 * 128;         // 16*192
  _Float16* x1ws = x0ws + 16 * 192;         // 2048*256

  auto rp = [&](const float* W, _Float16* wT, int K, int N) {
    int tot = K * N;
    k_repackT<<<(tot + 255) / 256, 256, 0, stream>>>(W, wT, K, N);
  };
  rp(W0a, w0aT, 192, 128);
  rp(W1a, w1aT, 256, 128);
  rp(W2a, w2aT, 256, 128);
  rp(W0b, w0bT, 128, 64);
  rp(W1b, w1bT, 128, 64);
  rp(W2b, w2bT, 128, 64);

  k_build_x0<<<16, 64, 0, stream>>>(f0, f1, x0ws);
  k_build_x1<<<2048, 64, 0, stream>>>(f0, f1, f2, x1ws);

  // order-0: 16 rows  -> 1 task  (MT=1)
  k_mlp_ws<6, 1><<<1, 256, 0, stream>>>(x0ws, w0aT, b0a, w0bT, b0b, out0, 1);
  // order-1: 2048 rows -> 64 tasks of 32 rows (MT=2)
  k_mlp_ws<8, 2><<<8, 256, 0, stream>>>(x1ws, w1aT, b1a, w1bT, b1b, out1, 64);
  // order-2: 262144 rows / 256 per block
  k_order2<<<1024, 256, 0, stream>>>(f1, f2, w2aT, b2a, w2bT, b2b, out2);
}
